// CombinedModel_sink400_13408887898553
// MI455X (gfx1250) — matrix operand order repaired and checked by reading, not compiled, not run on hardware
//
#include <hip/hip_runtime.h>
#include <hip/hip_bf16.h>
#include <cstdint>
#include <cstddef>

typedef __attribute__((ext_vector_type(16))) __bf16 v16bf;
typedef __attribute__((ext_vector_type(8)))  __bf16 v8bf;
typedef __attribute__((ext_vector_type(8)))  float  v8f;

namespace cfg {
constexpr int B = 32, N = 400, T = 315, D = 512, H = 256;
constexpr int TP  = 320;   // T padded to multiple of 16
constexpr int NKP = 416;   // N padded to multiple of 32 (used as WMMA K dim)
constexpr int ITERS = 100;
}

__device__ __forceinline__ float waveSum(float v) {
#pragma unroll
  for (int off = 16; off > 0; off >>= 1) v += __shfl_xor(v, off, 32);
  return v;
}

// ---------------- CDNA5 wave32 WMMA fragment loaders (bf16, 16x16x32) -------
// A 16x32: lane m=L&15; lanes<16 hold K 0..7 & 16..23, lanes>=16 hold K 8..15 & 24..31
__device__ __forceinline__ v16bf loadFragA(const __bf16* tileRow0, int ld, int k0, int lane) {
  const __bf16* p = tileRow0 + (size_t)(lane & 15) * ld + k0 + ((lane >> 4) << 3);
  v16bf f;
  *reinterpret_cast<v8bf*>(&f)       = *reinterpret_cast<const v8bf*>(p);
  *(reinterpret_cast<v8bf*>(&f) + 1) = *reinterpret_cast<const v8bf*>(p + 16);
  return f;
}
// B 32x16 (loaded from row-major B^T): lane n=L&15; lanes<16 hold K 0..7 & 16..23, lanes>=16 hold K 8..15 & 24..31
__device__ __forceinline__ v16bf loadFragB(const __bf16* tileRow0, int ld, int k0, int lane) {
  const __bf16* p = tileRow0 + (size_t)(lane & 15) * ld + k0 + ((lane >> 4) << 3);
  v16bf f;
  *reinterpret_cast<v8bf*>(&f)       = *reinterpret_cast<const v8bf*>(p);
  *(reinterpret_cast<v8bf*>(&f) + 1) = *reinterpret_cast<const v8bf*>(p + 16);
  return f;
}

// ---------------- Generic batched WMMA GEMM: C = A(bf16)[MxK] * Bt(bf16)[NxK]^T
// One wave per 16x16 output tile. All dims pre-padded -> full tiles, EXEC all ones.
__global__ void __launch_bounds__(32)
k_gemm(const __bf16* A, long sA, int lda,
       const __bf16* Bt, long sB, int ldb,
       float* Cf, __bf16* Cb, long sC, int ldc,
       int tilesN, int K) {
  const int lane = threadIdx.x;
  const int tm = blockIdx.x / tilesN;
  const int tn = blockIdx.x - tm * tilesN;
  const int b  = blockIdx.y;
  const __bf16* Ab = A  + (size_t)b * sA + (size_t)(tm * 16) * lda;
  const __bf16* Bb = Bt + (size_t)b * sB + (size_t)(tn * 16) * ldb;
  v8f acc = {};
  for (int k0 = 0; k0 < K; k0 += 32) {
    v16bf fa = loadFragA(Ab, lda, k0, lane);
    v16bf fb = loadFragB(Bb, ldb, k0, lane);
    acc = __builtin_amdgcn_wmma_f32_16x16x32_bf16(false, fa, false, fb,
                                                  (short)0, acc, false, false);
  }
  // C/D layout: VGPR r, lane L -> row = r + 8*(L>>4), col = L&15
  const int col = lane & 15;
  const int r0  = (lane >> 4) << 3;
  if (Cf) {
    float* o = Cf + (size_t)b * sC + (size_t)(tm * 16) * ldc + tn * 16 + col;
#pragma unroll
    for (int r = 0; r < 8; ++r) o[(size_t)(r0 + r) * ldc] = acc[r];
  } else {
    __bf16* o = Cb + (size_t)b * sC + (size_t)(tm * 16) * ldc + tn * 16 + col;
#pragma unroll
    for (int r = 0; r < 8; ++r) o[(size_t)(r0 + r) * ldc] = (__bf16)acc[r];
  }
}

// ---------------- f32 -> bf16 convert with optional transpose + zero padding
// dst[b][r][c], c<ldd; valid region (r<validR && c<validC), else 0.
__global__ void k_cvt(const float* src, long sSrc, int ldsrc, int trans,
                      __bf16* dst, long sDst, int R, int ldd,
                      int validR, int validC, int total) {
  int i = blockIdx.x * 256 + threadIdx.x;
  if (i >= total) return;
  int c = i % ldd;
  int j = i / ldd;
  int r = j % R;
  int b = j / R;
  float v = 0.f;
  if (r < validR && c < validC) {
    size_t si = (size_t)b * sSrc +
                (trans ? ((size_t)c * ldsrc + r) : ((size_t)r * ldsrc + c));
    v = src[si];
  }
  dst[(size_t)b * sDst + (size_t)r * ldd + c] = (__bf16)v;
}

// ---------------- InstanceNorm + 100 Sinkhorn iterations, one WG per batch --
__global__ void __launch_bounds__(1024)
k_sinkhorn(float* s, const float* gamma, const float* beta) {
  using namespace cfg;
  __shared__ float rowLSE[N];
  __shared__ float colLSE[TP];
  __shared__ float redA[32], redB[32];
  const int tid = threadIdx.x, lane = tid & 31, w = tid >> 5;
  float* sb = s + (size_t)blockIdx.x * N * TP;

  // ---- InstanceNorm over the N x T valid region
  float ls = 0.f, lq = 0.f;
  for (int i = tid; i < N * TP; i += 1024) {
    int c = i % TP;
    if (c < T) { float x = sb[i]; ls += x; lq += x * x; }
  }
  ls = waveSum(ls); lq = waveSum(lq);
  if (lane == 0) { redA[w] = ls; redB[w] = lq; }
  __syncthreads();
  float ts = 0.f, tq = 0.f;
#pragma unroll
  for (int i = 0; i < 32; ++i) { ts += redA[i]; tq += redB[i]; }
  const float invC = 1.f / ((float)N * (float)T);
  float mean  = ts * invC;
  float var   = tq * invC - mean * mean;
  float scale = gamma[0] * rsqrtf(var + 1e-5f);
  float shift = beta[0] - mean * scale;
  for (int i = tid; i < N * TP; i += 1024) {
    int c = i % TP;
    if (c < T) sb[i] = sb[i] * scale + shift;
  }
  __syncthreads();

  // ---- Sinkhorn iterations (row LSE, then col LSE of row-normalized values)
  for (int it = 0; it < ITERS; ++it) {
    // row logsumexp: one wave per row, streaming (m, sum)
    for (int r = w; r < N; r += 32) {
      const float* rp = sb + (size_t)r * TP;
      float m = -__builtin_huge_valf(), sum = 0.f;
      for (int c = lane; c < T; c += 32) {
        float x  = rp[c];
        float nm = fmaxf(m, x);
        sum = sum * __expf(m - nm) + __expf(x - nm);
        m = nm;
      }
#pragma unroll
      for (int off = 16; off > 0; off >>= 1) {
        float om = __shfl_xor(m, off, 32);
        float os = __shfl_xor(sum, off, 32);
        float nm = fmaxf(m, om);
        sum = sum * __expf(m - nm) + os * __expf(om - nm);
        m = nm;
      }
      if (lane == 0) rowLSE[r] = m + __logf(sum);
    }
    __syncthreads();
    // col logsumexp of (s - rowLSE): one thread per column, coalesced rows
    if (tid < T) {
      float m = -__builtin_huge_valf(), sum = 0.f;
      for (int r = 0; r < N; ++r) {
        float x  = sb[(size_t)r * TP + tid] - rowLSE[r];
        float nm = fmaxf(m, x);
        sum = sum * __expf(m - nm) + __expf(x - nm);
        m = nm;
      }
      colLSE[tid] = m + __logf(sum);
    }
    __syncthreads();
    // apply both normalizations
    for (int i = tid; i < N * TP; i += 1024) {
      int r = i / TP, c = i - r * TP;
      if (c < T) sb[i] -= rowLSE[r] + colLSE[c];
    }
    __syncthreads();
  }
}

// ---------------- exp + transpose: log_s [B][N][TP] -> s1t bf16 [B][TP][NKP]
__global__ void k_exp_tr(const float* s, __bf16* s1t) {
  using namespace cfg;
  __shared__ float tile[32][33];
  const int b  = blockIdx.z;
  const int n0 = blockIdx.x * 32;  // over NKP
  const int t0 = blockIdx.y * 32;  // over TP
  const int tx = threadIdx.x, ty = threadIdx.y;
  const float* sbp = s + (size_t)b * N * TP;
#pragma unroll
  for (int j = 0; j < 4; ++j) {
    int n = n0 + ty + j * 8;
    int t = t0 + tx;
    float v = 0.f;
    if (n < N && t < T) v = __expf(sbp[(size_t)n * TP + t]);
    tile[ty + j * 8][tx] = v;
  }
  __syncthreads();
  __bf16* ob = s1t + (size_t)b * TP * NKP;
#pragma unroll
  for (int j = 0; j < 4; ++j) {
    int t = t0 + ty + j * 8;
    int n = n0 + tx;
    ob[(size_t)t * NKP + n] = (__bf16)tile[tx][ty + j * 8];
  }
}

// ---------------- final head: pred = relu(h0 + txh + bh) @ Wo + bo ----------
__global__ void k_final(const float* h0, const float* txh, const float* bh,
                        const float* Wo, const float* bo, float* out) {
  using namespace cfg;
  const int lane = threadIdx.x & 31, w = threadIdx.x >> 5;
  const int row = blockIdx.x * 8 + w;     // over B*T, one wave per (b,t)
  if (row >= B * T) return;
  const int b = row / T, t = row - b * T;
  const float* hp = h0  + ((size_t)b * TP + t) * H;
  const float* tp = txh + (size_t)t * H;
  float acc = 0.f;
#pragma unroll
  for (int i = 0; i < H / 32; ++i) {
    int h = lane + i * 32;
    float v = hp[h] + tp[h] + bh[h];
    acc += fmaxf(v, 0.f) * Wo[h];
  }
  acc = waveSum(acc);
  if (lane == 0) out[row] = acc + bo[0];
}

// ======================= launch ============================================
extern "C" void kernel_launch(void* const* d_in, const int* in_sizes, int n_in,
                              void* d_out, int out_size, void* d_ws, size_t ws_size,
                              hipStream_t stream) {
  using namespace cfg;
  const float* features = (const float*)d_in[0];  // [B,N,D]
  const float* text     = (const float*)d_in[1];  // [T,D]
  const float* Amat     = (const float*)d_in[2];  // [D,D]
  const float* gamma    = (const float*)d_in[3];
  const float* beta     = (const float*)d_in[4];
  const float* Wh       = (const float*)d_in[5];  // [2D,H]
  const float* bh       = (const float*)d_in[6];  // [H]
  const float* Wo       = (const float*)d_in[7];  // [H,1]
  const float* bo       = (const float*)d_in[8];
  float* pred = (float*)d_out;                    // [B,T]
  (void)in_sizes; (void)n_in; (void)out_size; (void)ws_size;

  char* ws = (char*)d_ws;
  size_t off = 0;
  auto take = [&](size_t bytes) -> void* {
    void* p = ws + off; off += (bytes + 255) & ~(size_t)255; return p;
  };
  __bf16* featbf = (__bf16*)take((size_t)B * N * D * 2);     // [B][N][D]
  __bf16* featT  = (__bf16*)take((size_t)B * D * NKP * 2);   // [B][D][NKP] (n-pad 0)
  __bf16* textbf = (__bf16*)take((size_t)TP * D * 2);        // [TP][D] (t-pad 0)
  __bf16* Abf    = (__bf16*)take((size_t)D * D * 2);         // [D][D]
  __bf16* WhT    = (__bf16*)take((size_t)H * 2 * D * 2);     // [H][2D] transposed
  __bf16* tAbf   = (__bf16*)take((size_t)TP * D * 2);        // text @ A^T, bf16
  float*  smat   = (float*)take((size_t)B * N * TP * 4);     // sinkhorn state
  float*  txh    = (float*)take((size_t)TP * H * 4);         // text @ Wh[D:]
  __bf16* s1t    = (__bf16*)take((size_t)B * TP * NKP * 2);  // exp(log_s)^T
  __bf16* f1bf   = (__bf16*)take((size_t)B * TP * D * 2);    // features1, bf16
  float*  h0     = (float*)take((size_t)B * TP * H * 4);     // features1 @ Wh[:D]

  // ---- convert / pad / transpose inputs to bf16
  {
    int tot = B * N * D;   // featbf (row-major, no pad)
    k_cvt<<<dim3((tot + 255) / 256), 256, 0, stream>>>(
        features, 0, D, 0, featbf, 0, B * N, D, B * N, D, tot);
    tot = B * D * NKP;     // featT: per-batch transpose, n-cols padded to 416
    k_cvt<<<dim3((tot + 255) / 256), 256, 0, stream>>>(
        features, (long)N * D, D, 1, featT, (long)D * NKP, D, NKP, D, N, tot);
    tot = TP * D;          // text padded to 320 rows
    k_cvt<<<dim3((tot + 255) / 256), 256, 0, stream>>>(
        text, 0, D, 0, textbf, 0, TP, D, T, D, tot);
    tot = D * D;           // A row-major (acts as B^T for tA GEMM)
    k_cvt<<<dim3((tot + 255) / 256), 256, 0, stream>>>(
        Amat, 0, D, 0, Abf, 0, D, D, D, D, tot);
    tot = H * 2 * D;       // Wh transposed -> [H][2D]
    k_cvt<<<dim3((tot + 255) / 256), 256, 0, stream>>>(
        Wh, 0, H, 1, WhT, 0, H, 2 * D, H, 2 * D, tot);
  }

  // ---- tA[t,d] = sum_e text[t,e] * A[d,e]   (M=320,N=512,K=512) -> bf16
  k_gemm<<<dim3((TP / 16) * (D / 16), 1), 32, 0, stream>>>(
      textbf, 0, D, Abf, 0, D, nullptr, tAbf, 0, D, D / 16, D);

  // ---- s[b,n,t] = features[b,n,:] . tA[t,:]  (M=400,N=320,K=512) -> f32
  k_gemm<<<dim3((N / 16) * (TP / 16), B), 32, 0, stream>>>(
      featbf, (long)N * D, D, tAbf, 0, D,
      smat, nullptr, (long)N * TP, TP, TP / 16, D);

  // ---- txh[t,h] = text[t,:] @ Wh[D:,:]       (M=320,N=256,K=512) -> f32
  k_gemm<<<dim3((TP / 16) * (H / 16), 1), 32, 0, stream>>>(
      textbf, 0, D, WhT + D, 0, 2 * D, txh, nullptr, 0, H, H / 16, D);

  // ---- InstanceNorm + 100 Sinkhorn iterations (one WG per batch, L2-resident)
  k_sinkhorn<<<dim3(B), 1024, 0, stream>>>(smat, gamma, beta);

  // ---- s1t[b,t,n] = exp(log_s[b,n,t]), bf16, zero-padded
  k_exp_tr<<<dim3(NKP / 32, TP / 32, B), dim3(32, 8), 0, stream>>>(smat, s1t);

  // ---- features1[b,t,d] = sum_n s1t[t,n] * featT[d,n]  (M=320,N=512,K=416) -> bf16
  k_gemm<<<dim3((TP / 16) * (D / 16), B), 32, 0, stream>>>(
      s1t, (long)TP * NKP, NKP, featT, (long)D * NKP, NKP,
      nullptr, f1bf, (long)TP * D, D, D / 16, NKP);

  // ---- h0[b,t,h] = features1[b,t,:] @ Wh[:D,:]  (M=320,N=256,K=512) -> f32
  k_gemm<<<dim3((TP / 16) * (H / 16), B), 32, 0, stream>>>(
      f1bf, (long)TP * D, D, WhT, 0, 2 * D,
      h0, nullptr, (long)TP * H, H, H / 16, D);

  // ---- pred[b,t] = relu(h0 + txh + bh) @ Wo + bo
  k_final<<<dim3((B * T + 7) / 8), 256, 0, stream>>>(h0, txh, bh, Wo, bo, pred);
}
